// Decoder_56135222559417
// MI455X (gfx1250) — compile-verified, matrix operand order checked by reading, not run on hardware
//
#include <hip/hip_runtime.h>

// ---------------- problem constants ----------------
#define D_EMB 1024
#define NHEAD 16
#define DRED  64
#define BB    4
#define SS    1024
#define MTOT  (BB * SS)     // 4096 rows of activations
#define DFF   (4 * D_EMB)   // 4096

typedef unsigned int   u32;
typedef unsigned short u16;
typedef unsigned long long u64;
typedef u32   u32x4  __attribute__((ext_vector_type(4)));
typedef u32   u32x8  __attribute__((ext_vector_type(8)));
typedef float f32x8  __attribute__((ext_vector_type(8)));
typedef __bf16 bf16x16 __attribute__((ext_vector_type(16)));

// ---------------- small helpers ----------------
__device__ __forceinline__ u16 f2bf(float f) {
    union { float f; u32 u; } c; c.f = f;
    u32 u = c.u;
    u32 r = u + 0x7fffu + ((u >> 16) & 1u);   // round-to-nearest-even
    return (u16)(r >> 16);
}

__device__ __forceinline__ f32x8 zero8() {
    f32x8 z;
#pragma unroll
    for (int i = 0; i < 8; ++i) z[i] = 0.0f;
    return z;
}

// LDS byte offset of a shared-memory pointer: flat LDS addresses keep the
// offset in the low 32 bits.
__device__ __forceinline__ u32 lds_off(const void* p) {
    return (u32)(uintptr_t)p;
}

// Async global->LDS copy of 16 bytes per lane (GVS form, ASYNCcnt tracked).
__device__ __forceinline__ void async_copy_b128(u32 lds_byte, u32 goff_byte,
                                                const void* sbase) {
    asm volatile("global_load_async_to_lds_b128 %0, %1, %2"
                 :: "v"(lds_byte), "v"(goff_byte), "s"(sbase) : "memory");
}
__device__ __forceinline__ void wait_async0() {
    asm volatile("s_wait_asynccnt 0x0" ::: "memory");
}

// Load one 16-lane-striped bf16 fragment (A or B operand of 16x16x32 WMMA).
// rowbase points at the lane's matrix row (m for A, n for B); koff = kblock + half*8.
// Per ISA 7.12.2: VGPR0-3 hold K=[koff..koff+7], VGPR4-7 hold K=[koff+16..koff+23].
__device__ __forceinline__ bf16x16 load_frag(const u16* rowbase, int koff) {
    u32x4 lo = *reinterpret_cast<const u32x4*>(rowbase + koff);
    u32x4 hi = *reinterpret_cast<const u32x4*>(rowbase + koff + 16);
    u32x8 d;
    d[0] = lo[0]; d[1] = lo[1]; d[2] = lo[2]; d[3] = lo[3];
    d[4] = hi[0]; d[5] = hi[1]; d[6] = hi[2]; d[7] = hi[3];
    return __builtin_bit_cast(bf16x16, d);
}

__device__ __forceinline__ f32x8 wmma_bf16(bf16x16 a, bf16x16 b, f32x8 c) {
    return __builtin_amdgcn_wmma_f32_16x16x32_bf16(false, a, false, b, (short)0, c,
                                                   false, false);
}

// 16-lane (half-wave) reductions: rows of the WMMA C layout live across lanes
// 0-15 / 16-31, so xor masks 1,2,4,8 reduce exactly one matrix row.
__device__ __forceinline__ float hmax16(float x) {
#pragma unroll
    for (int m = 1; m < 16; m <<= 1) x = fmaxf(x, __shfl_xor(x, m, 32));
    return x;
}
__device__ __forceinline__ float hsum16(float x) {
#pragma unroll
    for (int m = 1; m < 16; m <<= 1) x += __shfl_xor(x, m, 32);
    return x;
}

// ---------------- cast / repack kernels ----------------
__global__ void k_cast_bf16(const float* __restrict__ in, u16* __restrict__ out, int n) {
    int i = blockIdx.x * blockDim.x + threadIdx.x;
    int st = gridDim.x * blockDim.x;
    for (; i < n; i += st) out[i] = f2bf(in[i]);
}

// wq [H=16][K=1024][R=64] -> Wt [N=H*R=1024][K=1024] bf16, n = h*64 + r
__global__ void k_pack_qkv(const float* __restrict__ w, u16* __restrict__ out) {
    int i = blockIdx.x * blockDim.x + threadIdx.x;
    int st = gridDim.x * blockDim.x;
    for (; i < 1024 * 1024; i += st) {
        int n = i >> 10, kk = i & 1023;
        out[i] = f2bf(w[(size_t)((n >> 6) * 1024 + kk) * 64 + (n & 63)]);
    }
}

// w [K][N] row-major -> Wt [N][K] bf16 (transpose + cast)
__global__ void k_pack_t(const float* __restrict__ w, u16* __restrict__ out, int N, int K) {
    int i = blockIdx.x * blockDim.x + threadIdx.x;
    int st = gridDim.x * blockDim.x;
    int tot = N * K;
    for (; i < tot; i += st) {
        int n = i / K, kk = i - n * K;
        out[i] = f2bf(w[(size_t)kk * N + n]);
    }
}

// ---------------- WMMA GEMM ----------------
// C[M,N] = act(A_bf16[M,K] x Wt_bf16[N,K]^T + bias) (+resid); writes f32 and/or bf16.
// Block tile 128x128, 8 waves (2 along M x 4 along N), wave tile 64x32 (4x2 WMMA tiles).
// K slabs staged to LDS with GLOBAL_LOAD_ASYNC_TO_LDS_B128 (no VGPR round-trip).
__global__ __launch_bounds__(256) void k_gemm(
    const u16* __restrict__ A, const u16* __restrict__ Wt,
    const float* __restrict__ bias, const float* __restrict__ resid,
    float* __restrict__ outF, u16* __restrict__ outB,
    int M, int N, int K, int relu)
{
    __shared__ u16 aS[128][72];   // row-pad 72: stride 36 dwords -> conflict-free b128
    __shared__ u16 bS[128][72];

    const int tid  = threadIdx.x;
    const int lane = tid & 31, l15 = lane & 15, hf = (lane >> 4) & 1;
    const int wid  = tid >> 5;
    const int bm = blockIdx.y * 128, bn = blockIdx.x * 128;
    const int wm = (wid & 1) * 64,   wn = (wid >> 1) * 32;

    f32x8 acc[4][2];
#pragma unroll
    for (int mt = 0; mt < 4; ++mt)
#pragma unroll
        for (int nt = 0; nt < 2; ++nt) acc[mt][nt] = zero8();

    for (int k0 = 0; k0 < K; k0 += 64) {
        __syncthreads();                      // prior slab fully consumed
#pragma unroll
        for (int j = 0; j < 4; ++j) {         // async-stage 128x64 of A and B
            int c0 = tid + j * 256;
            int r = c0 >> 3, c8 = (c0 & 7) << 3;
            u32 aoff = ((u32)(bm + r) * (u32)K + (u32)(k0 + c8)) * 2u;
            u32 boff = ((u32)(bn + r) * (u32)K + (u32)(k0 + c8)) * 2u;
            async_copy_b128(lds_off(&aS[r][c8]), aoff, A);
            async_copy_b128(lds_off(&bS[r][c8]), boff, Wt);
        }
        wait_async0();                        // this wave's copies landed in LDS
        __syncthreads();                      // everyone's copies visible
        if (k0 + 64 < K) {                    // global_prefetch next K slab
            __builtin_prefetch(A  + (size_t)(bm + (tid & 127)) * K + k0 + 64, 0, 1);
            __builtin_prefetch(Wt + (size_t)(bn + (tid & 127)) * K + k0 + 64, 0, 1);
        }
#pragma unroll
        for (int kb = 0; kb < 64; kb += 32) {
            bf16x16 af[4], bfr[2];
#pragma unroll
            for (int mt = 0; mt < 4; ++mt)
                af[mt] = load_frag(&aS[wm + mt * 16 + l15][0], kb + hf * 8);
#pragma unroll
            for (int nt = 0; nt < 2; ++nt)
                bfr[nt] = load_frag(&bS[wn + nt * 16 + l15][0], kb + hf * 8);
#pragma unroll
            for (int mt = 0; mt < 4; ++mt)
#pragma unroll
                for (int nt = 0; nt < 2; ++nt)
                    acc[mt][nt] = wmma_bf16(af[mt], bfr[nt], acc[mt][nt]);
        }
    }

#pragma unroll
    for (int mt = 0; mt < 4; ++mt)
#pragma unroll
        for (int nt = 0; nt < 2; ++nt)
#pragma unroll
            for (int v = 0; v < 8; ++v) {
                int row = bm + wm + mt * 16 + v + 8 * hf;   // C layout: M = v + 8*half
                int col = bn + wn + nt * 16 + l15;          //           N = lane&15
                float val = acc[mt][nt][v] + bias[col];
                if (relu)  val = fmaxf(val, 0.0f);
                if (resid) val += resid[(size_t)row * N + col];
                size_t o = (size_t)row * N + col;
                if (outF) outF[o] = val;
                if (outB) outB[o] = f2bf(val);
            }
}

// ---------------- fused flash attention (bf16 WMMA, online softmax) ----------------
// q,k,v,o: bf16 [B][S][H][R] ; one block = one (b,h) x 128 query rows; 8 waves x 16 rows.
// K chunk fetched by the Tensor Data Mover (2D tile DMA + LDS row padding);
// V chunk transposed manually (TDM has no transpose).
__global__ __launch_bounds__(256) void k_attn(
    const u16* __restrict__ qp, const u16* __restrict__ kp,
    const u16* __restrict__ vp, u16* __restrict__ op)
{
    __shared__ u16 kch[128][72];     // key chunk, row-major [t][r], TDM-padded rows
    __shared__ u16 vtS[64][136];     // value chunk transposed [r][t]
    __shared__ u16 pS[8][16][40];    // per-wave P transpose scratch

    const int tid  = threadIdx.x;
    const int lane = tid & 31, l15 = lane & 15, hf = (lane >> 4) & 1;
    const int wid  = tid >> 5;
    const int bx = blockIdx.x;
    const int qc = bx & 7;                 // S/128 = 8 chunks
    const int h  = (bx >> 3) & (NHEAD - 1);
    const int b  = bx >> 7;
    const int qbase = qc * 128 + wid * 16;

    // Q fragments straight from global (row is contiguous 64 bf16)
    const int srow = qbase + l15;
    const u16* qrow = qp + (size_t)((b * SS + srow) * NHEAD + h) * DRED;
    bf16x16 qf[2];
    qf[0] = load_frag(qrow, 0  + hf * 8);
    qf[1] = load_frag(qrow, 32 + hf * 8);

    f32x8 oacc[4];
#pragma unroll
    for (int rt = 0; rt < 4; ++rt) oacc[rt] = zero8();
    float run_m[8], run_l[8];
#pragma unroll
    for (int i = 0; i < 8; ++i) { run_m[i] = -1.0e30f; run_l[i] = 0.0f; }

    const u32 kch_lds = lds_off(&kch[0][0]);

    for (int nc = 0; nc < SS / 128; ++nc) {
        __syncthreads();                      // previous chunk fully consumed

        if (wid == 0) {
            // ---- TDM: DMA 128x64-element bf16 tile (row stride H*R) into kch ----
            u64 gaddr = (u64)(uintptr_t)kp +
                        2ull * (u64)(((b * SS + nc * 128) * NHEAD + h) * DRED);
            u32x4 g0;
            g0[0] = 1u;                                   // count=1, user descriptor
            g0[1] = kch_lds;                              // lds_addr
            g0[2] = (u32)(gaddr & 0xffffffffu);           // global_addr[31:0]
            g0[3] = (u32)((gaddr >> 32) & 0x01ffffffu)    // global_addr[56:32]
                    | (2u << 30);                         // type=2 ("image")
            u32x8 g1;
            g1[0] = (1u << 16)                            // data_size = 2 bytes
                  | (1u << 20)                            // pad_enable
                  | (4u << 22)                            // pad_interval: 32 DWORDs (128B row)
                  | (3u << 25);                           // pad_amount: 4 DWORDs (16B pad)
            g1[1] = (64u << 16);                          // tensor_dim0 = 64 (low 16)
            g1[2] = 0u;                                   // dim0 hi / dim1 low
            g1[3] = 16u | (64u << 16);                    // tensor_dim1 = 1<<20; tile_dim0 = 64
            g1[4] = 128u;                                 // tile_dim1 = 128, tile_dim2 = 0
            g1[5] = (u32)(NHEAD * DRED);                  // tensor_dim0_stride = 1024 elems
            g1[6] = 0u;
            g1[7] = 0u;
            asm volatile("tensor_load_to_lds %0, %1" :: "s"(g0), "s"(g1) : "memory");
            __builtin_amdgcn_s_wait_tensorcnt(0);
        }

        // ---- all threads: stage V transposed into vtS ----
#pragma unroll
        for (int j = 0; j < 4; ++j) {
            int c0 = tid + j * 256;
            int t = c0 >> 3, c8 = (c0 & 7) << 3;
            size_t gbase = (size_t)((b * SS + nc * 128 + t) * NHEAD + h) * DRED + c8;
            u32x4 vv = *reinterpret_cast<const u32x4*>(vp + gbase);
#pragma unroll
            for (int e = 0; e < 4; ++e) {
                vtS[c8 + 2 * e    ][t] = (u16)(vv[e] & 0xffffu);
                vtS[c8 + 2 * e + 1][t] = (u16)(vv[e] >> 16);
            }
        }
        __syncthreads();

#pragma unroll
        for (int tt = 0; tt < 4; ++tt) {        // 32 keys per step
            f32x8 s0 = zero8(), s1 = zero8();
#pragma unroll
            for (int kb = 0; kb < 2; ++kb) {    // contract over R=64
                bf16x16 b0 = load_frag(&kch[tt * 32 +      l15][0], kb * 32 + hf * 8);
                bf16x16 b1 = load_frag(&kch[tt * 32 + 16 + l15][0], kb * 32 + hf * 8);
                s0 = wmma_bf16(qf[kb], b0, s0);
                s1 = wmma_bf16(qf[kb], b1, s1);
            }
            float p0[8], p1[8], corr[8];
#pragma unroll
            for (int v = 0; v < 8; ++v) {       // online softmax, row = v + 8*half
                float a = s0[v] * 0.125f, c = s1[v] * 0.125f;   // 1/sqrt(64)
                float rm = hmax16(fmaxf(a, c));
                float nm = fmaxf(run_m[v], rm);
                float cr = __expf(run_m[v] - nm);
                float e0 = __expf(a - nm), e1 = __expf(c - nm);
                run_l[v] = run_l[v] * cr + hsum16(e0 + e1);
                run_m[v] = nm;
                corr[v] = cr; p0[v] = e0; p1[v] = e1;
            }
#pragma unroll
            for (int rt = 0; rt < 4; ++rt)
#pragma unroll
                for (int v = 0; v < 8; ++v) oacc[rt][v] *= corr[v];

            // transpose P (C layout -> A fragment) through per-wave LDS scratch
#pragma unroll
            for (int v = 0; v < 8; ++v) {
                int mrow = v + 8 * hf;
                pS[wid][mrow][l15]      = f2bf(p0[v]);
                pS[wid][mrow][l15 + 16] = f2bf(p1[v]);
            }
            __builtin_amdgcn_wave_barrier();    // keep DS store/load order (in-order DS unit)
            bf16x16 pf = load_frag(&pS[wid][l15][0], hf * 8);
#pragma unroll
            for (int rt = 0; rt < 4; ++rt) {
                bf16x16 vf = load_frag(&vtS[rt * 16 + l15][0], tt * 32 + hf * 8);
                oacc[rt] = wmma_bf16(pf, vf, oacc[rt]);
            }
        }
    }

#pragma unroll
    for (int v = 0; v < 8; ++v) {
        float inv = 1.0f / run_l[v];
        int mrow = v + 8 * hf;
        size_t base = (size_t)((b * SS + qbase + mrow) * NHEAD + h) * DRED;
#pragma unroll
        for (int rt = 0; rt < 4; ++rt)
            op[base + rt * 16 + l15] = f2bf(oacc[rt][v] * inv);
    }
}

// ---------------- LayerNorm ----------------
__global__ __launch_bounds__(256) void k_ln(
    const float* __restrict__ in, const float* __restrict__ g, const float* __restrict__ bb,
    float* __restrict__ outF, u16* __restrict__ outB)
{
    __shared__ float sh1[8], sh2[8];
    const int row = blockIdx.x, tid = threadIdx.x;
    const float* r = in + (size_t)row * D_EMB;
    float x[4], s1 = 0.0f, s2 = 0.0f;
#pragma unroll
    for (int i = 0; i < 4; ++i) {
        x[i] = r[tid + i * 256];
        s1 += x[i]; s2 += x[i] * x[i];
    }
#pragma unroll
    for (int m = 1; m < 32; m <<= 1) {
        s1 += __shfl_xor(s1, m, 32);
        s2 += __shfl_xor(s2, m, 32);
    }
    if ((tid & 31) == 0) { sh1[tid >> 5] = s1; sh2[tid >> 5] = s2; }
    __syncthreads();
    float t1 = 0.0f, t2 = 0.0f;
#pragma unroll
    for (int j = 0; j < 8; ++j) { t1 += sh1[j]; t2 += sh2[j]; }
    float mean = t1 * (1.0f / D_EMB);
    float var  = t2 * (1.0f / D_EMB) - mean * mean;
    float sc = rsqrtf(var + 1e-3f);
#pragma unroll
    for (int i = 0; i < 4; ++i) {
        int c = tid + i * 256;
        float y = (x[i] - mean) * sc * g[c] + bb[c];
        size_t o = (size_t)row * D_EMB + c;
        if (outF) outF[o] = y;
        if (outB) outB[o] = f2bf(y);
    }
}

// ---------------- host orchestration ----------------
extern "C" void kernel_launch(void* const* d_in, const int* in_sizes, int n_in,
                              void* d_out, int out_size, void* d_ws, size_t ws_size,
                              hipStream_t stream) {
    (void)in_sizes; (void)n_in; (void)out_size; (void)ws_size;

    const float* x      = (const float*)d_in[0];
    const float* ctx    = (const float*)d_in[1];
    const float* sa_wq  = (const float*)d_in[2];
    const float* sa_bq  = (const float*)d_in[3];
    const float* sa_wk  = (const float*)d_in[4];
    const float* sa_bk  = (const float*)d_in[5];
    const float* sa_wv  = (const float*)d_in[6];
    const float* sa_bv  = (const float*)d_in[7];
    const float* sa_wo  = (const float*)d_in[8];
    const float* sa_bo  = (const float*)d_in[9];
    const float* ca_wq  = (const float*)d_in[10];
    const float* ca_bq  = (const float*)d_in[11];
    const float* ca_wk  = (const float*)d_in[12];
    const float* ca_bk  = (const float*)d_in[13];
    const float* ca_wv  = (const float*)d_in[14];
    const float* ca_bv  = (const float*)d_in[15];
    const float* ca_wo  = (const float*)d_in[16];
    const float* ca_bo  = (const float*)d_in[17];
    const float* ln1_g  = (const float*)d_in[18];
    const float* ln1_b  = (const float*)d_in[19];
    const float* ln2_g  = (const float*)d_in[20];
    const float* ln2_b  = (const float*)d_in[21];
    const float* ln3_g  = (const float*)d_in[22];
    const float* ln3_b  = (const float*)d_in[23];
    const float* ffn_w1 = (const float*)d_in[24];
    const float* ffn_b1 = (const float*)d_in[25];
    const float* ffn_w2 = (const float*)d_in[26];
    const float* ffn_b2 = (const float*)d_in[27];

    char* ws = (char*)d_ws;
    size_t off = 0;
    auto alloc = [&](size_t bytes) -> void* {
        void* p = ws + off;
        off = (off + bytes + 255) & ~(size_t)255;
        return p;
    };

    const size_t ACT2 = (size_t)MTOT * D_EMB * 2;   // bf16 activation
    const size_t ACT4 = (size_t)MTOT * D_EMB * 4;   // f32 activation
    const size_t W2   = (size_t)D_EMB * D_EMB * 2;  // bf16 1024x1024 weight

    u16* xb     = (u16*)alloc(ACT2);
    u16* cb     = (u16*)alloc(ACT2);
    u16* wt_saq = (u16*)alloc(W2);
    u16* wt_sak = (u16*)alloc(W2);
    u16* wt_sav = (u16*)alloc(W2);
    u16* wt_sao = (u16*)alloc(W2);
    u16* wt_caq = (u16*)alloc(W2);
    u16* wt_cak = (u16*)alloc(W2);
    u16* wt_cav = (u16*)alloc(W2);
    u16* wt_cao = (u16*)alloc(W2);
    u16* wt1    = (u16*)alloc((size_t)DFF * D_EMB * 2);
    u16* wt2    = (u16*)alloc((size_t)D_EMB * DFF * 2);
    u16* qb     = (u16*)alloc(ACT2);
    u16* kb     = (u16*)alloc(ACT2);
    u16* vb     = (u16*)alloc(ACT2);
    u16* ob     = (u16*)alloc(ACT2);
    float* sumb = (float*)alloc(ACT4);
    float* ln1f = (float*)alloc(ACT4);
    u16*   ln1b = (u16*)alloc(ACT2);
    float* ln2f = (float*)alloc(ACT4);
    u16*   ln2b = (u16*)alloc(ACT2);
    u16*   f1b  = (u16*)alloc((size_t)MTOT * DFF * 2);

    const dim3 blk(256);
    // casts + packs
    k_cast_bf16<<<2048, blk, 0, stream>>>(x,   xb, MTOT * D_EMB);
    k_cast_bf16<<<2048, blk, 0, stream>>>(ctx, cb, MTOT * D_EMB);
    k_pack_qkv<<<1024, blk, 0, stream>>>(sa_wq, wt_saq);
    k_pack_qkv<<<1024, blk, 0, stream>>>(sa_wk, wt_sak);
    k_pack_qkv<<<1024, blk, 0, stream>>>(sa_wv, wt_sav);
    k_pack_t  <<<1024, blk, 0, stream>>>(sa_wo, wt_sao, D_EMB, D_EMB);
    k_pack_qkv<<<1024, blk, 0, stream>>>(ca_wq, wt_caq);
    k_pack_qkv<<<1024, blk, 0, stream>>>(ca_wk, wt_cak);
    k_pack_qkv<<<1024, blk, 0, stream>>>(ca_wv, wt_cav);
    k_pack_t  <<<1024, blk, 0, stream>>>(ca_wo, wt_cao, D_EMB, D_EMB);
    k_pack_t  <<<2048, blk, 0, stream>>>(ffn_w1, wt1, DFF, D_EMB);
    k_pack_t  <<<2048, blk, 0, stream>>>(ffn_w2, wt2, D_EMB, DFF);

    const dim3 gD(D_EMB / 128, MTOT / 128);   // N=1024 GEMMs
    const dim3 gF(DFF   / 128, MTOT / 128);   // N=4096 GEMM

    // ---- self attention ----
    k_gemm<<<gD, blk, 0, stream>>>(xb, wt_saq, sa_bq, nullptr, nullptr, qb, MTOT, D_EMB, D_EMB, 0);
    k_gemm<<<gD, blk, 0, stream>>>(xb, wt_sak, sa_bk, nullptr, nullptr, kb, MTOT, D_EMB, D_EMB, 0);
    k_gemm<<<gD, blk, 0, stream>>>(xb, wt_sav, sa_bv, nullptr, nullptr, vb, MTOT, D_EMB, D_EMB, 0);
    k_attn<<<BB * NHEAD * (SS / 128), blk, 0, stream>>>(qb, kb, vb, ob);
    k_gemm<<<gD, blk, 0, stream>>>(ob, wt_sao, sa_bo, x, sumb, nullptr, MTOT, D_EMB, D_EMB, 0);
    k_ln<<<MTOT, blk, 0, stream>>>(sumb, ln1_g, ln1_b, ln1f, ln1b);

    // ---- cross attention ----
    k_gemm<<<gD, blk, 0, stream>>>(ln1b, wt_caq, ca_bq, nullptr, nullptr, qb, MTOT, D_EMB, D_EMB, 0);
    k_gemm<<<gD, blk, 0, stream>>>(cb,   wt_cak, ca_bk, nullptr, nullptr, kb, MTOT, D_EMB, D_EMB, 0);
    k_gemm<<<gD, blk, 0, stream>>>(cb,   wt_cav, ca_bv, nullptr, nullptr, vb, MTOT, D_EMB, D_EMB, 0);
    k_attn<<<BB * NHEAD * (SS / 128), blk, 0, stream>>>(qb, kb, vb, ob);
    k_gemm<<<gD, blk, 0, stream>>>(ob, wt_cao, ca_bo, ln1f, sumb, nullptr, MTOT, D_EMB, D_EMB, 0);
    k_ln<<<MTOT, blk, 0, stream>>>(sumb, ln2_g, ln2_b, ln2f, ln2b);

    // ---- FFN ----
    k_gemm<<<gF, blk, 0, stream>>>(ln2b, wt1, ffn_b1, nullptr, nullptr, f1b, MTOT, DFF, D_EMB, 1);
    k_gemm<<<gD, blk, 0, stream>>>(f1b,  wt2, ffn_b2, ln2f, sumb, nullptr, MTOT, D_EMB, DFF, 0);
    k_ln<<<MTOT, blk, 0, stream>>>(sumb, ln3_g, ln3_b, (float*)d_out, nullptr);
}
